// RecurrentLaplaceSummarizer_34557306863844
// MI455X (gfx1250) — matrix operand order repaired and checked by reading, not compiled, not run on hardware
//
#include <hip/hip_runtime.h>
#include <hip/hip_bf16.h>
#include <math.h>

// ---------------------------------------------------------------------------
// Types
// ---------------------------------------------------------------------------
typedef __bf16 bf16;
typedef __attribute__((ext_vector_type(16))) __bf16 bf16x16;
typedef __attribute__((ext_vector_type(8)))  __bf16 bf16x8;
typedef __attribute__((ext_vector_type(8)))  float  f32x8;

#define B_    16
#define T_    2048
#define N_    64
#define D_    16
#define H_    1024
#define HEADS_ 8
#define HD_   128
#define LQ_   128
#define KP_   16          // laplace poles
#define BT_   (B_ * T_)   // 32768
#define LN10000 9.210340371976184f

__device__ __forceinline__ int lane_id() { return threadIdx.x & 31; }

__device__ __forceinline__ f32x8 zero8() {
  f32x8 z = {0.f, 0.f, 0.f, 0.f, 0.f, 0.f, 0.f, 0.f};
  return z;
}

__device__ __forceinline__ f32x8 wmma_bf16(bf16x16 a, bf16x16 b, f32x8 c) {
  // D = A(16x32) * B(32x16) + C, f32 accumulate
  return __builtin_amdgcn_wmma_f32_16x16x32_bf16(false, a, false, b,
                                                 (short)0, c, false, false);
}

// A-fragment (16x32 bf16). ISA layout: lane<16 -> row=lane, K={k0..k0+7, k0+16..k0+23}
//                                      lane>=16 -> row=lane-16, K={k0+8..15, k0+24..31}
__device__ __forceinline__ bf16x16 frag_a(const bf16* base, int stride) {
  int l = lane_id();
  const bf16* p = base + (l & 15) * stride + ((l >> 4) << 3);
  bf16x8 lo = *(const bf16x8*)p;
  bf16x8 hi = *(const bf16x8*)(p + 16);
  return __builtin_shufflevector(lo, hi, 0, 1, 2, 3, 4, 5, 6, 7,
                                 8, 9, 10, 11, 12, 13, 14, 15);
}

// B-fragment (32x16 bf16), source stored row-major as [col][k] (i.e. W[N][K] tile).
// ISA layout: lane<16 -> col=lane, K={0..7, 16..23}; lane>=16 -> col=lane-16, K={8..15, 24..31}.
__device__ __forceinline__ bf16x16 frag_b(const bf16* base, int stride) {
  int l = lane_id();
  const bf16* p = base + (l & 15) * stride + ((l >> 4) << 3);
  bf16x8 lo = *(const bf16x8*)p;
  bf16x8 hi = *(const bf16x8*)(p + 16);
  return __builtin_shufflevector(lo, hi, 0, 1, 2, 3, 4, 5, 6, 7,
                                 8, 9, 10, 11, 12, 13, 14, 15);
}

// ---------------------------------------------------------------------------
// f32 -> bf16 cast
// ---------------------------------------------------------------------------
__global__ void cast_kernel(const float* __restrict__ s, bf16* __restrict__ d,
                            size_t n) {
  size_t i = (size_t)blockIdx.x * 256 + threadIdx.x;
  if (i < n) d[i] = (bf16)s[i];
}

// ---------------------------------------------------------------------------
// tv heads: per (b,t,n) LayerNorm(16) -> 16x32 GELU MLP -> scalar.
// Also row means over N for the num_head stats.
// ---------------------------------------------------------------------------
__device__ __forceinline__ float tv_eval(const float* xv,
                                         const float* __restrict__ ln_g,
                                         const float* __restrict__ ln_b,
                                         const float* __restrict__ w1,
                                         const float* __restrict__ b1,
                                         const float* __restrict__ w2,
                                         const float* __restrict__ b2) {
  float mu = 0.f;
#pragma unroll
  for (int q = 0; q < D_; ++q) mu += xv[q];
  mu *= (1.f / D_);
  float var = 0.f;
#pragma unroll
  for (int q = 0; q < D_; ++q) { float dd = xv[q] - mu; var += dd * dd; }
  var *= (1.f / D_);
  float inv = rsqrtf(var + 1e-5f);
  float nx[D_];
#pragma unroll
  for (int q = 0; q < D_; ++q) nx[q] = (xv[q] - mu) * inv * ln_g[q] + ln_b[q];
  float out = b2[0];
  for (int h = 0; h < 32; ++h) {
    float a = b1[h];
#pragma unroll
    for (int q = 0; q < D_; ++q) a += w1[h * D_ + q] * nx[q];
    float g = 0.5f * a * (1.f + erff(a * 0.70710678118654752f));  // exact GELU
    out += w2[h] * g;
  }
  return out;
}

__global__ __launch_bounds__(64) void tv_kernel(
    const float* __restrict__ x,
    const float* vg, const float* vb, const float* vw1, const float* vb1,
    const float* vw2, const float* vb2,
    const float* tg, const float* tb, const float* tw1, const float* tb1,
    const float* tw2, const float* tb2,
    float* __restrict__ Vsig, float* __restrict__ Tsig,
    float* __restrict__ Vmean, float* __restrict__ Tmean) {
  int bt = blockIdx.x;          // 0..BT-1
  int n = threadIdx.x;          // 0..63
  int t = bt & (T_ - 1);
  const float* xp = x + ((size_t)bt * N_ + n) * D_;
  const float* xq = (t == 0) ? xp : (xp - (size_t)N_ * D_);
  float xv[D_], xd[D_];
#pragma unroll
  for (int q = 0; q < D_; ++q) { xv[q] = xp[q]; xd[q] = xv[q] - xq[q]; }
  float vs = tv_eval(xv, vg, vb, vw1, vb1, vw2, vb2);
  float ts = tv_eval(xd, tg, tb, tw1, tb1, tw2, tb2);
  Vsig[(size_t)bt * N_ + n] = vs;
  Tsig[(size_t)bt * N_ + n] = ts;
  __shared__ float rv[64], rt[64];
  rv[n] = vs; rt[n] = ts;
  __syncthreads();
  for (int o = 32; o > 0; o >>= 1) {
    if (n < o) { rv[n] += rv[n + o]; rt[n] += rt[n + o]; }
    __syncthreads();
  }
  if (n == 0) { Vmean[bt] = rv[0] * (1.f / N_); Tmean[bt] = rt[0] * (1.f / N_); }
}

// ---------------------------------------------------------------------------
// Laplace drive  u = sig @ wu^T + bu  (layout [2][B*T][K])
// ---------------------------------------------------------------------------
__global__ void u_kernel(const float* __restrict__ Tsig,
                         const float* __restrict__ Vsig,
                         const float* __restrict__ wu,
                         const float* __restrict__ bu,
                         float* __restrict__ U) {
  int idx = blockIdx.x * 256 + threadIdx.x;     // 2*BT*K = 1048576
  int k = idx & (KP_ - 1);
  int bt = (idx >> 4) & (BT_ - 1);
  int s = idx >> 19;
  const float* sig = (s ? Vsig : Tsig) + (size_t)bt * N_;
  float a = bu[k];
#pragma unroll 8
  for (int q = 0; q < N_; ++q) a += sig[q] * wu[k * N_ + q];
  U[idx] = a;
}

// ---------------------------------------------------------------------------
// Recurrent pole scan (T sequential), thread per (sig,b,k)
// ---------------------------------------------------------------------------
__global__ __launch_bounds__(32) void scan_kernel(
    const float* __restrict__ U, const float* __restrict__ s_real,
    const float* __restrict__ s_imag, const float* __restrict__ tau_raw,
    float* __restrict__ LT, float* __restrict__ LV) {
  int k = threadIdx.x;
  if (k >= KP_) return;
  int s = blockIdx.x >> 4, b = blockIdx.x & 15;
  float tau = log1pf(__expf(tau_raw[0])) + 1e-3f;   // softplus
  float alpha = fmaxf(s_real[k], 0.01f);
  float decay = __expf(-alpha * tau);
  float cth = __cosf(s_imag[k] * tau);
  float sth = __sinf(s_imag[k] * tau);
  const float* u = U + ((size_t)s * BT_ + (size_t)b * T_) * KP_ + k;
  float* out = (s ? LV : LT) + (size_t)b * T_ * (2 * KP_) + 2 * k;
  float pr = 0.f, pi = 0.f;
  for (int t = 0; t < T_; ++t) {
    float ut = u[(size_t)t * KP_];
    float xr = decay * (cth * pr - sth * pi) + ut;
    float xi = decay * (sth * pr + cth * pi);
    out[(size_t)t * (2 * KP_)] = xr;
    out[(size_t)t * (2 * KP_) + 1] = xi;
    pr = xr; pi = xi;
  }
}

// ---------------------------------------------------------------------------
// num_head coeff MLP: stats(4) -> LN -> 8 GELU -> 4
// ---------------------------------------------------------------------------
__global__ void coeff_kernel(const float* __restrict__ Tmean,
                             const float* __restrict__ Vmean,
                             const float* g, const float* b,
                             const float* w1, const float* b1,
                             const float* w2, const float* b2,
                             float* __restrict__ coeff) {
  int bt = blockIdx.x * 256 + threadIdx.x;
  if (bt >= BT_) return;
  int t = bt & (T_ - 1);
  float s[4];
  s[0] = Tmean[bt];
  s[1] = Vmean[bt];
  s[2] = t ? (s[0] - Tmean[bt - 1]) : 0.f;
  s[3] = t ? (s[1] - Vmean[bt - 1]) : 0.f;
  float mu = 0.25f * (s[0] + s[1] + s[2] + s[3]);
  float var = 0.f;
#pragma unroll
  for (int q = 0; q < 4; ++q) { float dd = s[q] - mu; var += dd * dd; }
  var *= 0.25f;
  float inv = rsqrtf(var + 1e-5f);
  float nx[4];
#pragma unroll
  for (int q = 0; q < 4; ++q) nx[q] = (s[q] - mu) * inv * g[q] + b[q];
  float hq[8];
#pragma unroll
  for (int h = 0; h < 8; ++h) {
    float a = b1[h];
#pragma unroll
    for (int q = 0; q < 4; ++q) a += w1[h * 4 + q] * nx[q];
    hq[h] = 0.5f * a * (1.f + erff(a * 0.70710678118654752f));
  }
#pragma unroll
  for (int o = 0; o < 4; ++o) {
    float a = b2[o];
#pragma unroll
    for (int h = 0; h < 8; ++h) a += w2[o * 8 + h] * hq[h];
    coeff[(size_t)bt * 4 + o] = a;
  }
}

// ---------------------------------------------------------------------------
// Combine: polewise diff + L = a0*LT + a1*LdT + b0*LV + b1*LdV, plus attn bias
// ---------------------------------------------------------------------------
__global__ void combine_kernel(const float* __restrict__ LT,
                               const float* __restrict__ LV,
                               const float* __restrict__ coeff,
                               const float* __restrict__ s_real,
                               const float* __restrict__ s_imag,
                               const float* __restrict__ tau_raw,
                               const float* __restrict__ residual,
                               const float* __restrict__ bias_w,
                               const float* __restrict__ bias_b,
                               float* __restrict__ Lf,
                               bf16* __restrict__ Lbf,
                               float* __restrict__ bias8) {
  int bt = blockIdx.x * 256 + threadIdx.x;
  if (bt >= BT_) return;
  float tau = log1pf(__expf(tau_raw[0])) + 1e-3f;
  float a0 = coeff[(size_t)bt * 4 + 0], a1 = coeff[(size_t)bt * 4 + 1];
  float b0 = coeff[(size_t)bt * 4 + 2], b1 = coeff[(size_t)bt * 4 + 3];
  float Lloc[2 * KP_];
#pragma unroll
  for (int k = 0; k < KP_; ++k) {
    float a = fmaxf(s_real[k], 0.01f), w = s_imag[k];
    float A00 = -a * tau + residual[k * 4 + 0];
    float A01 = -w * tau + residual[k * 4 + 1];
    float A10 =  w * tau + residual[k * 4 + 2];
    float A11 = -a * tau + residual[k * 4 + 3];
    size_t base = (size_t)bt * (2 * KP_) + 2 * k;
    float ltr = LT[base], lti = LT[base + 1];
    float lvr = LV[base], lvi = LV[base + 1];
    float ldtr = A00 * ltr + A01 * lti, ldti = A10 * ltr + A11 * lti;
    float ldvr = A00 * lvr + A01 * lvi, ldvi = A10 * lvr + A11 * lvi;
    Lloc[2 * k]     = a0 * ltr + a1 * ldtr + b0 * lvr + b1 * ldvr;
    Lloc[2 * k + 1] = a0 * lti + a1 * ldti + b0 * lvi + b1 * ldvi;
  }
#pragma unroll
  for (int j = 0; j < 2 * KP_; ++j) {
    Lf[(size_t)bt * 32 + j] = Lloc[j];
    Lbf[(size_t)bt * 32 + j] = (bf16)Lloc[j];
  }
#pragma unroll
  for (int h = 0; h < HEADS_; ++h) {
    float a = bias_b[h];
#pragma unroll
    for (int j = 0; j < 32; ++j) a += Lloc[j] * bias_w[h * 32 + j];
    bias8[(size_t)bt * HEADS_ + h] = a;
  }
}

// ---------------------------------------------------------------------------
// Generic bf16 WMMA GEMM: C[M,N] = A[M,K] @ W[N,K]^T (+bias) with epilogues.
//  MODE 0: bf16 out, row-major, +bias
//  MODE 1: ctx -> memory: +bias(ctx_b) +pos_emb, FiLM with L/film_w/film_b,
//          write bf16 memory rows [b*2T + t]
//  MODE 2: head-split bf16 out (k_/v_ projection), +bias
//  MODE 3: f32 out, +bias +queries residual (pre-LN)
// Block 256 thr = 8 waves; block tile 128x128, K-step 32; wave tile 64x32.
// ---------------------------------------------------------------------------
template <int MODE>
__global__ __launch_bounds__(256) void gemm_bf16_kernel(
    const bf16* __restrict__ A, const bf16* __restrict__ W,
    const float* __restrict__ bias, int Klen, int Ncols,
    float* __restrict__ outf, bf16* __restrict__ outb,
    const float* __restrict__ e0, const float* __restrict__ e1,
    const float* __restrict__ e2) {
  __shared__ __align__(16) bf16 ldsA[128 * 40];
  __shared__ __align__(16) bf16 ldsB[128 * 40];
  const int tid = threadIdx.x;
  const int row0 = blockIdx.y * 128;
  const int col0 = blockIdx.x * 128;
  const int wave = tid >> 5;
  const int wm = wave & 1, wn = wave >> 1;
  f32x8 acc[4][2];
#pragma unroll
  for (int i = 0; i < 4; ++i)
#pragma unroll
    for (int j = 0; j < 2; ++j) acc[i][j] = zero8();

  const int lr = tid >> 1;          // 0..127
  const int lc = (tid & 1) * 16;    // 0 or 16
  const bf16* Ap = A + (size_t)(row0 + lr) * Klen + lc;
  const bf16* Wp = W + (size_t)(col0 + lr) * Klen + lc;
  bf16* la = ldsA + lr * 40 + lc;
  bf16* lb = ldsB + lr * 40 + lc;

  for (int k0 = 0; k0 < Klen; k0 += 32) {
    bf16x8 a0 = *(const bf16x8*)(Ap);
    bf16x8 a1 = *(const bf16x8*)(Ap + 8);
    bf16x8 w0 = *(const bf16x8*)(Wp);
    bf16x8 w1 = *(const bf16x8*)(Wp + 8);
    Ap += 32; Wp += 32;
    if (k0 + 32 < Klen) {                       // global_prefetch_b8 next tiles
      __builtin_prefetch(Ap, 0, 3);
      __builtin_prefetch(Wp, 0, 3);
    }
    __syncthreads();                            // previous-iter LDS reads done
    *(bf16x8*)la = a0; *(bf16x8*)(la + 8) = a1;
    *(bf16x8*)lb = w0; *(bf16x8*)(lb + 8) = w1;
    __syncthreads();
    bf16x16 bfr[2];
#pragma unroll
    for (int j = 0; j < 2; ++j)
      bfr[j] = frag_b(ldsB + (wn * 32 + j * 16) * 40, 40);
#pragma unroll
    for (int i = 0; i < 4; ++i) {
      bf16x16 afr = frag_a(ldsA + (wm * 64 + i * 16) * 40, 40);
#pragma unroll
      for (int j = 0; j < 2; ++j) acc[i][j] = wmma_bf16(afr, bfr[j], acc[i][j]);
    }
  }

  // ---- epilogue (C-layout: elem ii -> row = ii + 8*(lane>=16), col = lane&15)
  const int lane = tid & 31;
#pragma unroll
  for (int i = 0; i < 4; ++i) {
#pragma unroll
    for (int j = 0; j < 2; ++j) {
      const int tr0 = row0 + wm * 64 + i * 16;
      const int tc0 = col0 + wn * 32 + j * 16;
      const int col = tc0 + (lane & 15);
      const float bcol = bias[col];
#pragma unroll
      for (int ii = 0; ii < 8; ++ii) {
        const int row = tr0 + ii + ((lane >> 4) << 3);
        float v = acc[i][j][ii] + bcol;
        if (MODE == 0) {
          outb[(size_t)row * Ncols + col] = (bf16)v;
        } else if (MODE == 1) {
          const int t = row & (T_ - 1);
          const int bb = row >> 11;
          float inv = __expf(-(float)(col & ~1) * (LN10000 / (float)H_));
          float ang = (float)t * inv;
          float pe = (col & 1) ? __cosf(ang) : __sinf(ang);
          float kv = v + pe;
          const float* Lr = e0 + (size_t)row * 32;
          const float* fg = e1 + (size_t)col * 32;
          const float* fb = e1 + (size_t)(H_ + col) * 32;
          float gm = e2[col], be = e2[H_ + col];
#pragma unroll 8
          for (int q = 0; q < 32; ++q) {
            gm += Lr[q] * fg[q];
            be += Lr[q] * fb[q];
          }
          size_t mrow = (size_t)bb * (2 * T_) + t;
          outb[mrow * H_ + col] = (bf16)((1.f + gm) * kv + be);
        } else if (MODE == 2) {
          const int bb = row >> 12;           // M = B * 2T
          const int m = row & (2 * T_ - 1);
          const int head = col >> 7;
          const int off = col & (HD_ - 1);
          outb[(((size_t)bb * HEADS_ + head) * (2 * T_) + m) * HD_ + off] = (bf16)v;
        } else {  // MODE 3
          outf[(size_t)row * Ncols + col] =
              v + e2[(size_t)(row & (LQ_ - 1)) * Ncols + col];
        }
      }
    }
  }
}

// ---------------------------------------------------------------------------
// lap_tok = L @ tok_w^T + tok_b -> memory rows [b*2T + T + t], one WMMA/tile
// ---------------------------------------------------------------------------
__global__ __launch_bounds__(256) void tok_kernel(const bf16* __restrict__ Lbf,
                                                  const bf16* __restrict__ tokw,
                                                  const float* __restrict__ tokb,
                                                  bf16* __restrict__ memb) {
  int wg = blockIdx.x * 8 + (threadIdx.x >> 5);  // global wave id
  int tm = wg >> 6;                              // 2048 row tiles
  int tn = wg & 63;                              // 64 col tiles
  int row0 = tm * 16, col0 = tn * 16;
  bf16x16 af = frag_a(Lbf + (size_t)row0 * 32, 32);
  bf16x16 bf = frag_b(tokw + (size_t)col0 * 32, 32);
  f32x8 acc = zero8();
  acc = wmma_bf16(af, bf, acc);
  int lane = lane_id();
  int col = col0 + (lane & 15);
  float bb = tokb[col];
#pragma unroll
  for (int ii = 0; ii < 8; ++ii) {
    int row = row0 + ii + ((lane >> 4) << 3);
    int b = row >> 11, t = row & (T_ - 1);
    memb[((size_t)b * (2 * T_) + T_ + t) * H_ + col] = (bf16)(acc[ii] + bb);
  }
}

// ---------------------------------------------------------------------------
// Flash attention: one workgroup per (b,head); 8 waves x 16 query rows.
// Keys/values streamed in 32-key blocks; online softmax; all matmuls WMMA.
// ---------------------------------------------------------------------------
__global__ __launch_bounds__(256) void attn_kernel(
    const bf16* __restrict__ qbf,     // [LQ, H]
    const bf16* __restrict__ kbuf,    // [B, heads, 2T, hd]
    const bf16* __restrict__ vbuf,    // [B, heads, 2T, hd]
    const float* __restrict__ bias8,  // [B*T, heads]
    bf16* __restrict__ attnout) {     // [B*LQ, H]
  __shared__ __align__(16) bf16 ldsK[32 * 136];
  __shared__ __align__(16) bf16 ldsVt[128 * 40];
  __shared__ __align__(16) bf16 ldsP[8 * 16 * 40];
  const int tid = threadIdx.x;
  const int wave = tid >> 5;
  const int lane = tid & 31;
  const int b = blockIdx.x >> 3;
  const int h = blockIdx.x & 7;

  bf16x16 qf[4];
  const bf16* qb = qbf + (size_t)(wave * 16) * H_ + h * HD_;
#pragma unroll
  for (int c = 0; c < 4; ++c) qf[c] = frag_a(qb + c * 32, H_);

  f32x8 oacc[8];
#pragma unroll
  for (int j = 0; j < 8; ++j) oacc[j] = zero8();
  float m[8], l[8];
#pragma unroll
  for (int i = 0; i < 8; ++i) { m[i] = -1e30f; l[i] = 0.f; }

  const bf16* kb = kbuf + ((size_t)b * HEADS_ + h) * (2 * T_) * HD_;
  const bf16* vb = vbuf + ((size_t)b * HEADS_ + h) * (2 * T_) * HD_;
  const float scale = 0.08838834764831845f;  // 1/sqrt(128)
  bf16* pp_base = ldsP + wave * 16 * 40;

  for (int key0 = 0; key0 < 2 * T_; key0 += 32) {
    __syncthreads();  // all waves done with previous K/V tiles
    {               // cooperative load: K row-major [32][136], V transposed [128][40]
      int r = tid >> 3;
      int cc = (tid & 7) * 16;
      const bf16* ks = kb + (size_t)(key0 + r) * HD_ + cc;
      bf16x8 k0v = *(const bf16x8*)ks;
      bf16x8 k1v = *(const bf16x8*)(ks + 8);
      *(bf16x8*)(ldsK + r * 136 + cc) = k0v;
      *(bf16x8*)(ldsK + r * 136 + cc + 8) = k1v;
      const bf16* vs = vb + (size_t)(key0 + r) * HD_ + cc;
      bf16x8 v0 = *(const bf16x8*)vs;
      bf16x8 v1 = *(const bf16x8*)(vs + 8);
#pragma unroll
      for (int jj = 0; jj < 8; ++jj) {
        ldsVt[(cc + jj) * 40 + r] = v0[jj];
        ldsVt[(cc + 8 + jj) * 40 + r] = v1[jj];
      }
    }
    __syncthreads();

    // scores: q(16x128) @ K^T -> two 16x16 tiles (keys key0..+15, key0+16..+31)
    f32x8 s0 = zero8(), s1 = zero8();
#pragma unroll
    for (int c = 0; c < 4; ++c) {
      bf16x16 kf0 = frag_b(ldsK + c * 32, 136);
      bf16x16 kf1 = frag_b(ldsK + 16 * 136 + c * 32, 136);
      s0 = wmma_bf16(qf[c], kf0, s0);
      s1 = wmma_bf16(qf[c], kf1, s1);
    }
    int c0 = key0 + (lane & 15);
    int c1 = key0 + 16 + (lane & 15);
    float bias0 = (c0 < T_) ? bias8[((size_t)b * T_ + c0) * HEADS_ + h] : 0.f;
    float bias1 = (c1 < T_) ? bias8[((size_t)b * T_ + c1) * HEADS_ + h] : 0.f;

#pragma unroll
    for (int i = 0; i < 8; ++i) {
      float e0v = s0[i] * scale + bias0;
      float e1v = s1[i] * scale + bias1;
      float mx = fmaxf(e0v, e1v);
#pragma unroll
      for (int o = 1; o < 16; o <<= 1) mx = fmaxf(mx, __shfl_xor(mx, o, 32));
      float mn = fmaxf(m[i], mx);
      float corr = __expf(m[i] - mn);
      m[i] = mn;
      float p0 = __expf(e0v - mn);
      float p1 = __expf(e1v - mn);
      float rs = p0 + p1;
#pragma unroll
      for (int o = 1; o < 16; o <<= 1) rs += __shfl_xor(rs, o, 32);
      l[i] = l[i] * corr + rs;
#pragma unroll
      for (int j = 0; j < 8; ++j) oacc[j][i] *= corr;
      int prow = i + ((lane >> 4) << 3);
      bf16* pp = pp_base + prow * 40;
      pp[lane & 15] = (bf16)p0;
      pp[16 + (lane & 15)] = (bf16)p1;
    }
    // wave-private LDS transpose: make sure all lane stores landed
    asm volatile("s_wait_dscnt 0" ::: "memory");
    bf16x16 pf = frag_a(pp_base, 40);
#pragma unroll
    for (int j = 0; j < 8; ++j) {
      bf16x16 vf = frag_b(ldsVt + (j * 16) * 40, 40);
      oacc[j] = wmma_bf16(pf, vf, oacc[j]);
    }
  }

  // epilogue: normalize, write per-head slice of [B*LQ, H]
#pragma unroll
  for (int i = 0; i < 8; ++i) {
    float invl = 1.f / l[i];
    int row = wave * 16 + i + ((lane >> 4) << 3);
#pragma unroll
    for (int j = 0; j < 8; ++j) {
      int col = h * HD_ + j * 16 + (lane & 15);
      attnout[((size_t)b * LQ_ + row) * H_ + col] = (bf16)(oacc[j][i] * invl);
    }
  }
}

// ---------------------------------------------------------------------------
// Final LayerNorm over H per row -> d_out
// ---------------------------------------------------------------------------
__global__ __launch_bounds__(256) void ln_kernel(const float* __restrict__ pre,
                                                 const float* __restrict__ g,
                                                 const float* __restrict__ be,
                                                 float* __restrict__ out) {
  int row = blockIdx.x;
  const float* p = pre + (size_t)row * H_;
  __shared__ float red[256];
  float s = 0.f;
  for (int c = threadIdx.x; c < H_; c += 256) s += p[c];
  red[threadIdx.x] = s;
  __syncthreads();
  for (int o = 128; o > 0; o >>= 1) {
    if (threadIdx.x < o) red[threadIdx.x] += red[threadIdx.x + o];
    __syncthreads();
  }
  float mu = red[0] * (1.f / H_);
  __syncthreads();
  float v = 0.f;
  for (int c = threadIdx.x; c < H_; c += 256) {
    float dd = p[c] - mu;
    v += dd * dd;
  }
  red[threadIdx.x] = v;
  __syncthreads();
  for (int o = 128; o > 0; o >>= 1) {
    if (threadIdx.x < o) red[threadIdx.x] += red[threadIdx.x + o];
    __syncthreads();
  }
  float inv = rsqrtf(red[0] * (1.f / H_) + 1e-5f);
  for (int c = threadIdx.x; c < H_; c += 256)
    out[(size_t)row * H_ + c] = (p[c] - mu) * inv * g[c] + be[c];
}

// ---------------------------------------------------------------------------
// Host launcher
// ---------------------------------------------------------------------------
extern "C" void kernel_launch(void* const* d_in, const int* in_sizes, int n_in,
                              void* d_out, int out_size, void* d_ws,
                              size_t ws_size, hipStream_t stream) {
  (void)in_sizes; (void)n_in; (void)out_size; (void)ws_size;
  const float* x         = (const float*)d_in[0];
  const float* ctx_w     = (const float*)d_in[1];
  const float* ctx_b     = (const float*)d_in[2];
  const float* v_ln_g    = (const float*)d_in[3];
  const float* v_ln_b    = (const float*)d_in[4];
  const float* v_w1      = (const float*)d_in[5];
  const float* v_b1      = (const float*)d_in[6];
  const float* v_w2      = (const float*)d_in[7];
  const float* v_b2      = (const float*)d_in[8];
  const float* t_ln_g    = (const float*)d_in[9];
  const float* t_ln_b    = (const float*)d_in[10];
  const float* t_w1      = (const float*)d_in[11];
  const float* t_b1      = (const float*)d_in[12];
  const float* t_w2      = (const float*)d_in[13];
  const float* t_b2      = (const float*)d_in[14];
  const float* nh_ln_g   = (const float*)d_in[15];
  const float* nh_ln_b   = (const float*)d_in[16];
  const float* nh_w1     = (const float*)d_in[17];
  const float* nh_b1     = (const float*)d_in[18];
  const float* nh_w2     = (const float*)d_in[19];
  const float* nh_b2     = (const float*)d_in[20];
  const float* s_real    = (const float*)d_in[21];
  const float* s_imag    = (const float*)d_in[22];
  const float* tau_raw   = (const float*)d_in[23];
  const float* wu        = (const float*)d_in[24];
  const float* bu        = (const float*)d_in[25];
  const float* residual  = (const float*)d_in[26];
  const float* film_w    = (const float*)d_in[27];
  const float* film_b    = (const float*)d_in[28];
  const float* bias_w    = (const float*)d_in[29];
  const float* bias_b    = (const float*)d_in[30];
  const float* tok_w     = (const float*)d_in[31];
  const float* tok_b     = (const float*)d_in[32];
  const float* queries   = (const float*)d_in[33];
  const float* in_proj_w = (const float*)d_in[34];
  const float* in_proj_b = (const float*)d_in[35];
  const float* out_proj_w= (const float*)d_in[36];
  const float* out_proj_b= (const float*)d_in[37];
  const float* norm_g    = (const float*)d_in[38];
  const float* norm_b    = (const float*)d_in[39];

  // ---- workspace carve (deterministic layout, 256B aligned) ----
  char* ws = (char*)d_ws;
  size_t off = 0;
  auto carve = [&](size_t bytes) -> char* {
    char* p = ws + off;
    off += (bytes + 255) & ~(size_t)255;
    return p;
  };
  const size_t NX = (size_t)B_ * T_ * N_ * D_;       // 33554432
  bf16*  xbf      = (bf16*)carve(NX * 2);
  bf16*  ctxw_bf  = (bf16*)carve((size_t)H_ * H_ * 2);
  bf16*  inprj_bf = (bf16*)carve((size_t)3 * H_ * H_ * 2);
  bf16*  outw_bf  = (bf16*)carve((size_t)H_ * H_ * 2);
  bf16*  tokw_bf  = (bf16*)carve((size_t)H_ * 32 * 2);
  bf16*  qin_bf   = (bf16*)carve((size_t)LQ_ * H_ * 2);
  float* Tsig     = (float*)carve((size_t)BT_ * N_ * 4);
  float* Vsig     = (float*)carve((size_t)BT_ * N_ * 4);
  float* Tmean    = (float*)carve((size_t)BT_ * 4);
  float* Vmean    = (float*)carve((size_t)BT_ * 4);
  float* coeff    = (float*)carve((size_t)BT_ * 4 * 4);
  float* Ubuf     = (float*)carve((size_t)2 * BT_ * KP_ * 4);
  float* LTb      = (float*)carve((size_t)BT_ * 32 * 4);
  float* LVb      = (float*)carve((size_t)BT_ * 32 * 4);
  float* Lf32     = (float*)carve((size_t)BT_ * 32 * 4);
  bf16*  Lbf      = (bf16*)carve((size_t)BT_ * 32 * 2);
  float* bias8    = (float*)carve((size_t)BT_ * HEADS_ * 4);
  bf16*  memb     = (bf16*)carve((size_t)B_ * 2 * T_ * H_ * 2);
  bf16*  kbuf     = (bf16*)carve((size_t)B_ * 2 * T_ * H_ * 2);
  bf16*  vbuf     = (bf16*)carve((size_t)B_ * 2 * T_ * H_ * 2);
  bf16*  qproj    = (bf16*)carve((size_t)LQ_ * H_ * 2);
  bf16*  attnb    = (bf16*)carve((size_t)B_ * LQ_ * H_ * 2);
  float* preln    = (float*)carve((size_t)B_ * LQ_ * H_ * 4);

  auto cast = [&](const float* s, bf16* d, size_t n) {
    int blocks = (int)((n + 255) / 256);
    cast_kernel<<<blocks, 256, 0, stream>>>(s, d, n);
  };
  cast(x, xbf, NX);
  cast(ctx_w, ctxw_bf, (size_t)H_ * H_);
  cast(in_proj_w, inprj_bf, (size_t)3 * H_ * H_);
  cast(out_proj_w, outw_bf, (size_t)H_ * H_);
  cast(tok_w, tokw_bf, (size_t)H_ * 32);
  cast(queries, qin_bf, (size_t)LQ_ * H_);

  // tv heads + row means
  tv_kernel<<<BT_, 64, 0, stream>>>(x, v_ln_g, v_ln_b, v_w1, v_b1, v_w2, v_b2,
                                    t_ln_g, t_ln_b, t_w1, t_b1, t_w2, t_b2,
                                    Vsig, Tsig, Vmean, Tmean);
  // laplace drive + recurrent scan
  u_kernel<<<(2 * BT_ * KP_) / 256, 256, 0, stream>>>(Tsig, Vsig, wu, bu, Ubuf);
  scan_kernel<<<32, 32, 0, stream>>>(Ubuf, s_real, s_imag, tau_raw, LTb, LVb);
  // coefficients + combine
  coeff_kernel<<<BT_ / 256, 256, 0, stream>>>(Tmean, Vmean, nh_ln_g, nh_ln_b,
                                              nh_w1, nh_b1, nh_w2, nh_b2, coeff);
  combine_kernel<<<BT_ / 256, 256, 0, stream>>>(LTb, LVb, coeff, s_real, s_imag,
                                                tau_raw, residual, bias_w,
                                                bias_b, Lf32, Lbf, bias8);
  // ctx projection + pos-emb + fused FiLM -> memory rows [0,T)
  gemm_bf16_kernel<1><<<dim3(H_ / 128, BT_ / 128), 256, 0, stream>>>(
      xbf, ctxw_bf, ctx_b, H_, H_, nullptr, memb, Lf32, film_w, film_b);
  // lap_tok -> memory rows [T,2T)
  tok_kernel<<<(BT_ / 16) * (H_ / 16) / 8, 256, 0, stream>>>(Lbf, tokw_bf,
                                                             tok_b, memb);
  // k/v projections, head-split layout
  gemm_bf16_kernel<2><<<dim3(H_ / 128, (B_ * 2 * T_) / 128), 256, 0, stream>>>(
      memb, inprj_bf + (size_t)H_ * H_, in_proj_b + H_, H_, H_, nullptr, kbuf,
      nullptr, nullptr, nullptr);
  gemm_bf16_kernel<2><<<dim3(H_ / 128, (B_ * 2 * T_) / 128), 256, 0, stream>>>(
      memb, inprj_bf + (size_t)2 * H_ * H_, in_proj_b + 2 * H_, H_, H_, nullptr,
      vbuf, nullptr, nullptr, nullptr);
  // q projection (batch-independent)
  gemm_bf16_kernel<0><<<dim3(H_ / 128, LQ_ / 128), 256, 0, stream>>>(
      qin_bf, inprj_bf, in_proj_b, H_, H_, nullptr, qproj, nullptr, nullptr,
      nullptr);
  // flash attention, one WG per (b, head)
  attn_kernel<<<B_ * HEADS_, 256, 0, stream>>>(qproj, kbuf, vbuf, bias8, attnb);
  // out projection + residual (queries)
  gemm_bf16_kernel<3><<<dim3(H_ / 128, (B_ * LQ_) / 128), 256, 0, stream>>>(
      attnb, outw_bf, out_proj_b, H_, H_, preln, nullptr, nullptr, nullptr,
      queries);
  // final LayerNorm -> d_out (f32)
  ln_kernel<<<B_ * LQ_, 256, 0, stream>>>(preln, norm_g, norm_b, (float*)d_out);
}
